// Attention_58325655880209
// MI455X (gfx1250) — matrix operand order repaired and checked by reading, not compiled, not run on hardware
//
#include <hip/hip_runtime.h>

// ---------------------------------------------------------------------------
// Bahdanau attention for MI455X (gfx1250, wave32, WMMA).
// B=64, T=2048, D=512, U=512.
//   q  = h @ W1 + b1                      (tiny)
//   k  = E @ W2 + b2                      (68.7 GFLOP  -> split-bf16 WMMA)
//   lg = tanh(q+k) @ V + bV              (fused into GEMM epilogue)
//   a  = softmax_T(lg)
//   ctx= sum_t a * E                      (HBM bound, 268 MB)
// ---------------------------------------------------------------------------

typedef __attribute__((ext_vector_type(16))) __bf16          v16bf;
typedef __attribute__((ext_vector_type(8)))  float           v8f;
typedef __attribute__((ext_vector_type(16))) unsigned short  v16u16;
typedef __attribute__((ext_vector_type(4)))  float           v4f;

#define B_  64
#define T_  2048
#define D_  512
#define U_  512

// round-to-nearest-even fp32 -> bf16 (bit pattern)
__device__ __forceinline__ unsigned short bf16_rne(float f) {
  unsigned u = __float_as_uint(f);
  unsigned r = u + 0x7fffu + ((u >> 16) & 1u);
  return (unsigned short)(r >> 16);
}
__device__ __forceinline__ float bf16_to_f32(unsigned short h) {
  return __uint_as_float(((unsigned)h) << 16);
}

// ---------------------------------------------------------------------------
// Kernel 0: pack W2 [D,U] into WMMA B-operand bf16 hi/lo planes.
// Layout: PB[nt][ks][lane][e], nt=U-tile(32), ks=K-step(16), lane(32), e(16).
// B 32x16 bf16 striping (wave32): lane = n + 16*g holds column n, K = 8*g + e (e<8), 16 + 8*g + (e-8) (e>=8).
// ---------------------------------------------------------------------------
__global__ __launch_bounds__(256) void pack_w2_kernel(
    const float* __restrict__ W2,
    unsigned short* __restrict__ PBhi,
    unsigned short* __restrict__ PBlo) {
  int idx = blockIdx.x * blockDim.x + threadIdx.x;   // 0 .. 32*16*32-1
  if (idx >= 32 * 16 * 32) return;
  int lane = idx & 31;
  int ks   = (idx >> 5) & 15;
  int nt   = idx >> 9;
  int nl   = lane & 15;
  int g    = lane >> 4;
  int u    = nt * 16 + nl;
  size_t base = (size_t)idx * 16;
  #pragma unroll
  for (int e = 0; e < 16; ++e) {
    int d = ks * 32 + (e < 8 ? 8 * g + e : 16 + 8 * g + (e - 8));
    float w = W2[(size_t)d * U_ + u];
    unsigned short hb = bf16_rne(w);
    unsigned short lb = bf16_rne(w - bf16_to_f32(hb));
    PBhi[base + e] = hb;
    PBlo[base + e] = lb;
  }
}

// ---------------------------------------------------------------------------
// Kernel 1: qb[b][u] = sum_d h[b][d]*W1[d][u] + b1[u] + b2[u]
// ---------------------------------------------------------------------------
__global__ __launch_bounds__(256) void qproj_kernel(
    const float* __restrict__ H, const float* __restrict__ W1,
    const float* __restrict__ b1, const float* __restrict__ b2,
    float* __restrict__ qb) {
  __shared__ float sh[D_];
  int b = blockIdx.x;
  for (int i = threadIdx.x; i < D_; i += 256) sh[i] = H[(size_t)b * D_ + i];
  __syncthreads();
  int u0 = threadIdx.x, u1 = threadIdx.x + 256;
  float a0 = 0.f, a1 = 0.f;
  for (int d = 0; d < D_; ++d) {
    float h = sh[d];
    a0 = fmaf(h, W1[(size_t)d * U_ + u0], a0);
    a1 = fmaf(h, W1[(size_t)d * U_ + u1], a1);
  }
  qb[(size_t)b * U_ + u0] = a0 + b1[u0] + b2[u0];
  qb[(size_t)b * U_ + u1] = a1 + b1[u1] + b2[u1];
}

// ---------------------------------------------------------------------------
// Kernel 2: main fused kernel.
// Block = (b, 32 rows of T). 4 waves; wave w owns U-tiles nt = 8w..8w+7 and
// BOTH 16-row M-tiles (B operand reused across M). Split-bf16 (3 WMMA) GEMM,
// then fused tanh(q+k)*V reduction -> logits[b][t].
// A 16x32 bf16 striping (wave32): lane = m + 16*g holds row m,
//   elems e<8 -> K = 8g+e, e>=8 -> K = 16+8g+(e-8).
// ---------------------------------------------------------------------------
__global__ __launch_bounds__(128) void attn_logits_kernel(
    const float* __restrict__ E, const float* __restrict__ qb,
    const unsigned short* __restrict__ PBhi,
    const unsigned short* __restrict__ PBlo,
    const float* __restrict__ Vv, const float* __restrict__ bV,
    float* __restrict__ logits) {
  const int b    = blockIdx.y;
  const int t0   = blockIdx.x * 32;
  const int wave = threadIdx.x >> 5;
  const int lane = threadIdx.x & 31;
  const int nl   = lane & 15;
  const int g    = lane >> 4;

  const float* arow0 = E + ((size_t)b * T_ + (t0 + nl)) * D_;
  const float* arow1 = arow0 + (size_t)16 * D_;

  v8f acc0[8] = {};
  v8f acc1[8] = {};

  for (int ks = 0; ks < 16; ++ks) {
    const int d0 = ks * 32 + 8 * g;
    if (ks < 15) {
      __builtin_prefetch(arow0 + d0 + 32, 0, 0);
      __builtin_prefetch(arow1 + d0 + 32, 0, 0);
    }
    // two contiguous 8-float runs per M-tile row
    v4f x0 = *(const v4f*)(arow0 + d0);
    v4f x1 = *(const v4f*)(arow0 + d0 + 4);
    v4f x2 = *(const v4f*)(arow0 + d0 + 16);
    v4f x3 = *(const v4f*)(arow0 + d0 + 20);
    v4f y0 = *(const v4f*)(arow1 + d0);
    v4f y1 = *(const v4f*)(arow1 + d0 + 4);
    v4f y2 = *(const v4f*)(arow1 + d0 + 16);
    v4f y3 = *(const v4f*)(arow1 + d0 + 20);

    float a0[16], a1[16];
    #pragma unroll
    for (int i = 0; i < 4; ++i) {
      a0[i] = x0[i]; a0[4 + i] = x1[i]; a0[8 + i] = x2[i]; a0[12 + i] = x3[i];
      a1[i] = y0[i]; a1[4 + i] = y1[i]; a1[8 + i] = y2[i]; a1[12 + i] = y3[i];
    }
    v16u16 h0, l0, h1, l1;
    #pragma unroll
    for (int e = 0; e < 16; ++e) {
      unsigned short hb = bf16_rne(a0[e]);
      h0[e] = hb;  l0[e] = bf16_rne(a0[e] - bf16_to_f32(hb));
      hb = bf16_rne(a1[e]);
      h1[e] = hb;  l1[e] = bf16_rne(a1[e] - bf16_to_f32(hb));
    }
    v16bf A0h = __builtin_bit_cast(v16bf, h0);
    v16bf A0l = __builtin_bit_cast(v16bf, l0);
    v16bf A1h = __builtin_bit_cast(v16bf, h1);
    v16bf A1l = __builtin_bit_cast(v16bf, l1);

    #pragma unroll
    for (int j = 0; j < 8; ++j) {
      const int nt = wave * 8 + j;
      const size_t off = (((size_t)(nt * 16 + ks)) * 32 + lane) * 16;
      v16bf Bh = *(const v16bf*)(PBhi + off);
      v16bf Bl = *(const v16bf*)(PBlo + off);
      // split-fp32: acc += Ah*Bh + Ah*Bl + Al*Bh
      acc0[j] = __builtin_amdgcn_wmma_f32_16x16x32_bf16(false, A0h, false, Bh,
                                                        (short)0, acc0[j], false, false);
      acc0[j] = __builtin_amdgcn_wmma_f32_16x16x32_bf16(false, A0h, false, Bl,
                                                        (short)0, acc0[j], false, false);
      acc0[j] = __builtin_amdgcn_wmma_f32_16x16x32_bf16(false, A0l, false, Bh,
                                                        (short)0, acc0[j], false, false);
      acc1[j] = __builtin_amdgcn_wmma_f32_16x16x32_bf16(false, A1h, false, Bh,
                                                        (short)0, acc1[j], false, false);
      acc1[j] = __builtin_amdgcn_wmma_f32_16x16x32_bf16(false, A1h, false, Bl,
                                                        (short)0, acc1[j], false, false);
      acc1[j] = __builtin_amdgcn_wmma_f32_16x16x32_bf16(false, A1l, false, Bh,
                                                        (short)0, acc1[j], false, false);
    }
  }

  // Fused epilogue: logit contribution = sum_u tanh(k + qb) * V[u].
  // C/D layout: lane (nl,g), VGPR r -> element (M = r + 8g, N = nl).
  float rs0[8] = {}, rs1[8] = {};
  #pragma unroll
  for (int j = 0; j < 8; ++j) {
    const int u  = (wave * 8 + j) * 16 + nl;
    const float q  = qb[(size_t)b * U_ + u];
    const float vv = Vv[u];
    #pragma unroll
    for (int r = 0; r < 8; ++r) {
      rs0[r] += tanhf(acc0[j][r] + q) * vv;
      rs1[r] += tanhf(acc1[j][r] + q) * vv;
    }
  }
  // reduce over N within each 16-lane half (M differs between halves!)
  #pragma unroll
  for (int off = 1; off < 16; off <<= 1) {
    #pragma unroll
    for (int r = 0; r < 8; ++r) {
      rs0[r] += __shfl_xor(rs0[r], off, 32);
      rs1[r] += __shfl_xor(rs1[r], off, 32);
    }
  }
  __shared__ float s_logit[32];
  if (threadIdx.x < 32) s_logit[threadIdx.x] = 0.f;
  __syncthreads();
  if (nl == 0) {
    #pragma unroll
    for (int r = 0; r < 8; ++r) {
      atomicAdd(&s_logit[g * 8 + r], rs0[r]);        // rows t0 + (r+8g)
      atomicAdd(&s_logit[16 + g * 8 + r], rs1[r]);   // rows t0 + 16 + (r+8g)
    }
  }
  __syncthreads();
  if (threadIdx.x < 32)
    logits[(size_t)b * T_ + t0 + threadIdx.x] = s_logit[threadIdx.x] + bV[0];
}

// ---------------------------------------------------------------------------
// Kernel 3: softmax over T per batch row. Writes attention weights.
// ---------------------------------------------------------------------------
__global__ __launch_bounds__(256) void softmax_kernel(
    const float* __restrict__ logits, float* __restrict__ attn) {
  __shared__ float red[256];
  int b = blockIdx.x;
  const float* lg = logits + (size_t)b * T_;
  float v[8];
  float m = -3.0e38f;
  #pragma unroll
  for (int i = 0; i < 8; ++i) {
    v[i] = lg[threadIdx.x + 256 * i];
    m = fmaxf(m, v[i]);
  }
  red[threadIdx.x] = m; __syncthreads();
  for (int s = 128; s > 0; s >>= 1) {
    if (threadIdx.x < s) red[threadIdx.x] = fmaxf(red[threadIdx.x], red[threadIdx.x + s]);
    __syncthreads();
  }
  m = red[0]; __syncthreads();
  float sum = 0.f;
  #pragma unroll
  for (int i = 0; i < 8; ++i) { v[i] = expf(v[i] - m); sum += v[i]; }
  red[threadIdx.x] = sum; __syncthreads();
  for (int s = 128; s > 0; s >>= 1) {
    if (threadIdx.x < s) red[threadIdx.x] += red[threadIdx.x + s];
    __syncthreads();
  }
  float inv = 1.0f / red[0];
  #pragma unroll
  for (int i = 0; i < 8; ++i)
    attn[(size_t)b * T_ + threadIdx.x + 256 * i] = v[i] * inv;
}

// ---------------------------------------------------------------------------
// Kernel 4: context[b][d] = sum_t attn[b][t] * E[b][t][d]  (HBM bound)
// ---------------------------------------------------------------------------
__global__ __launch_bounds__(256) void context_kernel(
    const float* __restrict__ E, const float* __restrict__ attn,
    float* __restrict__ ctx) {
  __shared__ float sa[T_];
  int b = blockIdx.x;
  for (int i = threadIdx.x; i < T_; i += 256) sa[i] = attn[(size_t)b * T_ + i];
  __syncthreads();
  int d0 = threadIdx.x, d1 = threadIdx.x + 256;
  float a0 = 0.f, a1 = 0.f;
  const float* Eb = E + (size_t)b * T_ * D_;
  for (int t = 0; t < T_; ++t) {
    float w = sa[t];
    a0 = fmaf(w, Eb[(size_t)t * D_ + d0], a0);
    a1 = fmaf(w, Eb[(size_t)t * D_ + d1], a1);
  }
  ctx[(size_t)b * D_ + d0] = a0;
  ctx[(size_t)b * D_ + d1] = a1;
}

// ---------------------------------------------------------------------------
extern "C" void kernel_launch(void* const* d_in, const int* in_sizes, int n_in,
                              void* d_out, int out_size, void* d_ws, size_t ws_size,
                              hipStream_t stream) {
  const float* H  = (const float*)d_in[0];   // [B,D]
  const float* E  = (const float*)d_in[1];   // [B,T,D]
  const float* W1 = (const float*)d_in[2];   // [D,U]
  const float* b1 = (const float*)d_in[3];   // [U]
  const float* W2 = (const float*)d_in[4];   // [D,U]
  const float* b2 = (const float*)d_in[5];   // [U]
  const float* Vv = (const float*)d_in[6];   // [U,1]
  const float* bV = (const float*)d_in[7];   // [1]

  float* out  = (float*)d_out;
  float* ctx  = out;                 // [B,D]   = 32768 floats
  float* attn = out + B_ * D_;       // [B,T,1] = 131072 floats

  // workspace: qb(128KB) | logits(512KB) | PBhi(512KB) | PBlo(512KB)
  float* qb     = (float*)d_ws;
  float* logits = qb + B_ * U_;
  unsigned short* PBhi = (unsigned short*)(logits + (size_t)B_ * T_);
  unsigned short* PBlo = PBhi + (size_t)D_ * U_;

  pack_w2_kernel<<<(32 * 16 * 32) / 256, 256, 0, stream>>>(W2, PBhi, PBlo);
  qproj_kernel<<<B_, 256, 0, stream>>>(H, W1, b1, b2, qb);
  dim3 grid(T_ / 32, B_);
  attn_logits_kernel<<<grid, 128, 0, stream>>>(E, qb, PBhi, PBlo, Vv, bV, logits);
  softmax_kernel<<<B_, 256, 0, stream>>>(logits, attn);
  context_kernel<<<B_, 256, 0, stream>>>(E, attn, ctx);
}
